// SimpleGeoBloom_82214263980288
// MI455X (gfx1250) — matrix operand order repaired and checked by reading, not compiled, not run on hardware
//
#include <hip/hip_runtime.h>

typedef __bf16 bf16;
typedef __attribute__((ext_vector_type(16))) __bf16 v16bf;
typedef __attribute__((ext_vector_type(8)))  __bf16 v8bf;
typedef __attribute__((ext_vector_type(8)))  float   v8f;

#define BQ   32
#define BP   256
#define LTOK 16
#define CIN  8192
#define H1   256
#define H2   32
#define COUT 8192

static __device__ inline v16bf cat8(v8bf lo, v8bf hi) {
  return __builtin_shufflevector(lo, hi, 0,1,2,3,4,5,6,7,8,9,10,11,12,13,14,15);
}
static __device__ inline v8f wmma_bf16f32(v16bf a, v16bf b, v8f c) {
  return __builtin_amdgcn_wmma_f32_16x16x32_bf16(false, a, false, b, (short)0, c, false, false);
}
// A fragment (16x32 bf16, M x K): prow points at (row m, K-step base).
// element e<8 -> prow[hk*8+e]; e>=8 -> prow[16+hk*8+(e-8)]
static __device__ inline v16bf a_frag(const bf16* prow, int hk) {
  const bf16* p = prow + hk * 8;
  v8bf lo = *(const v8bf*)p;
  v8bf hi = *(const v8bf*)(p + 16);
  return cat8(lo, hi);
}
// B fragment (32x16 bf16, K x N) column-major: pcol points at (col n, K-step base),
// element e<8 -> pcol[hk*8+e]; e>=8 -> pcol[16+hk*8+(e-8)]
static __device__ inline v16bf b_frag(const bf16* pcol, int hk) {
  const bf16* p = pcol + hk * 8;
  v8bf lo = *(const v8bf*)p;
  v8bf hi = *(const v8bf*)(p + 16);
  return cat8(lo, hi);
}

// ---- CDNA5 async global->LDS copy (ASYNCcnt domain), portable via inline asm ----
// Copies 16 bytes per enabled lane: LDS[lds_off] = MEM[gptr]. Tracked by ASYNCcnt.
static __device__ inline void async_ld_b128(unsigned lds_off, const void* gptr) {
  asm volatile("global_load_async_to_lds_b128 %0, %1, off"
               :: "v"(lds_off), "v"(gptr)
               : "memory");
}
static __device__ inline void wait_async0() {
  asm volatile("s_wait_asynccnt 0x0" ::: "memory");
}

// ---------------- weight preconversion ----------------
__global__ __launch_bounds__(256) void k_f2bf(const float* __restrict__ in,
                                              bf16* __restrict__ out, int n) {
  int i = blockIdx.x * 256 + threadIdx.x;
  if (i < n) out[i] = (bf16)in[i];
}
// conv_w [oc][ic][5] f32 -> out[t][oc][ic] bf16 (K=ic contiguous for B frags)
__global__ __launch_bounds__(256) void k_convw(const float* __restrict__ w,
                                               bf16* __restrict__ out) {
  int i = blockIdx.x * 256 + threadIdx.x;
  if (i < 5 * H1 * H1) {
    int t = i / (H1 * H1);
    int rem = i % (H1 * H1);
    int oc = rem / H1, ic = rem % H1;
    out[i] = (bf16)w[(oc * H1 + ic) * 5 + t];
  }
}

// ---------------- encoder GEMM: Y = X @ enc_w^T ----------------
// Async double-buffered staging of X (f32) into LDS, bf16 conversion in LDS,
// 16 waves cover the full N=256 with one shared 16-row A tile.
__global__ __launch_bounds__(512) void k_encode(const float* __restrict__ X,
                                                const bf16* __restrict__ Wb,
                                                float* __restrict__ Yo) {
  __shared__ __align__(16) float sraw[2][16 * 512];  // 2 x 32KB raw f32 chunks
  __shared__ __align__(16) bf16  sA[16 * 512];       // 16KB bf16 A tile chunk
  int mt = blockIdx.x;
  int tid = threadIdx.x;
  int lane = tid & 31, wave = tid >> 5;
  int hk = lane >> 4, ml = lane & 15;
  int n0 = wave << 4;
  v8f acc = {};
  const float* Xt = X + (size_t)mt * 16 * CIN;
  const bf16* wcol = Wb + (size_t)(n0 + ml) * CIN;

  // this thread's 16-f32 (64B) staging slice within an 8192-element chunk
  const int sbase = tid * 16;
  const int sr = sbase >> 9, sk = sbase & 511;

  // prologue: issue async copies for chunk 0
  {
    const float* g = Xt + (size_t)sr * CIN + sk;
    unsigned l0 = (unsigned)(uintptr_t)&sraw[0][sbase];
    async_ld_b128(l0,      g);
    async_ld_b128(l0 + 16, g + 4);
    async_ld_b128(l0 + 32, g + 8);
    async_ld_b128(l0 + 48, g + 12);
  }
  for (int kc = 0; kc < CIN; kc += 512) {
    int cur = (kc >> 9) & 1;
    wait_async0();        // my async copies (issued last iter / prologue) landed
    __syncthreads();      // all waves' copies visible; sA free for rewrite
    // convert raw f32 chunk -> bf16 A tile
    for (int i = tid; i < 16 * 512; i += 512)
      sA[i] = (bf16)sraw[cur][i];
    // prefetch next chunk asynchronously while we convert + compute
    if (kc + 512 < CIN) {
      const float* g = Xt + (size_t)sr * CIN + (kc + 512) + sk;
      unsigned l0 = (unsigned)(uintptr_t)&sraw[cur ^ 1][sbase];
      async_ld_b128(l0,      g);
      async_ld_b128(l0 + 16, g + 4);
      async_ld_b128(l0 + 32, g + 8);
      async_ld_b128(l0 + 48, g + 12);
    }
    __syncthreads();      // sA ready
#pragma unroll 4
    for (int ks = 0; ks < 512; ks += 32) {
      v16bf a = a_frag(sA + ml * 512 + ks, hk);
      v16bf w = b_frag(wcol + kc + ks, hk);
      acc = wmma_bf16f32(a, w, acc);
    }
  }
  float* yp = Yo + (size_t)mt * 16 * H1 + n0 + ml;
#pragma unroll
  for (int r = 0; r < 8; ++r) {
    int m = r + hk * 8;
    yp[(size_t)m * H1] = acc[r];
  }
}

// ---------------- conv block (3 layers, k=5, pad=2) + residual + mean ----------------
__global__ __launch_bounds__(512) void k_conv(const float* __restrict__ Y,
                                              const bf16* __restrict__ Wt,
                                              const float* __restrict__ cb1,
                                              const float* __restrict__ cb2,
                                              const float* __restrict__ cb3,
                                              bf16* __restrict__ embq,
                                              bf16* __restrict__ embp) {
  __shared__ __align__(16) bf16 sbuf[2][20 * H1];  // rows -2..17 (pad rows 0,1,18,19)
  __shared__ float scolsum[H1];
  int b = blockIdx.x;
  int tid = threadIdx.x;
  int lane = tid & 31, wave = tid >> 5;
  int hk = lane >> 4, ml = lane & 15;
  int n0 = wave << 4;
  const float* Yb = Y + (size_t)b * LTOK * H1;
  // zero pad rows of both buffers
  for (int i = tid; i < 2 * 4 * H1; i += 512) {
    int bufi = i / (4 * H1);
    int rr = (i / H1) & 3;
    int c = i % H1;
    int row = (rr < 2) ? rr : rr + 16;
    sbuf[bufi][row * H1 + c] = (bf16)0.f;
  }
  // stage encoder output (bf16) + residual column sums (f32)
  for (int i = tid; i < LTOK * H1; i += 512) {
    int l = i / H1, c = i % H1;
    sbuf[0][(l + 2) * H1 + c] = (bf16)Yb[i];
  }
  if (tid < H1) {
    float s = 0.f;
    for (int l = 0; l < LTOK; ++l) s += Yb[l * H1 + tid];
    scolsum[tid] = s;
  }
  __syncthreads();
  for (int layer = 0; layer < 3; ++layer) {
    int cur = layer & 1;
    const float* bias = (layer == 0) ? cb1 : (layer == 1 ? cb2 : cb3);
    const bf16* Wl = Wt + (size_t)layer * 5 * H1 * H1;
    float bn = bias[n0 + ml];
    v8f acc;
#pragma unroll
    for (int r = 0; r < 8; ++r) acc[r] = bn;
    for (int t = 0; t < 5; ++t) {
      const bf16* arow = sbuf[cur] + (ml + t) * H1;  // logical row m+t-2 at LDS row m+t
      const bf16* wcol = Wl + ((size_t)(t * H1) + n0 + ml) * H1;
#pragma unroll
      for (int ks = 0; ks < H1; ks += 32) {
        v16bf a = a_frag(arow + ks, hk);
        v16bf w = b_frag(wcol + ks, hk);
        acc = wmma_bf16f32(a, w, acc);
      }
    }
#pragma unroll
    for (int r = 0; r < 8; ++r) acc[r] = fmaxf(acc[r], 0.f);
    if (layer < 2) {
#pragma unroll
      for (int r = 0; r < 8; ++r) {
        int m = r + hk * 8;
        sbuf[cur ^ 1][(m + 2) * H1 + n0 + ml] = (bf16)acc[r];
      }
      __syncthreads();
    } else {
      // mean over L of (conv_out + x), then relu for decode input
      float s = 0.f;
#pragma unroll
      for (int r = 0; r < 8; ++r) s += acc[r];
      s += __shfl_xor(s, 16);
      if (hk == 0) {
        int c = n0 + ml;
        float e = (s + scolsum[c]) * (1.f / 16.f);
        e = fmaxf(e, 0.f);
        if (b < BQ) embq[b * H1 + c] = (bf16)e;
        else        embp[(b - BQ) * H1 + c] = (bf16)e;
      }
    }
  }
}

// ---------------- blooms ----------------
__global__ __launch_bounds__(256) void k_bloom(const float* __restrict__ X,
                                               float* __restrict__ bloom) {
  int idx = blockIdx.x * 256 + threadIdx.x;   // over B*CIN
  int b = idx >> 13, c = idx & (CIN - 1);
  float s = 0.f;
  for (int l = 0; l < LTOK; ++l) s += X[((size_t)(b * LTOK + l)) * CIN + c];
  bloom[idx] = fminf(s, 1.f);
}

__global__ __launch_bounds__(256) void k_qbits(const float* __restrict__ qb,
                                               float* __restrict__ qbits) {
  __shared__ float red[256];
  int q = blockIdx.x;
  float s = 0.f;
  for (int c = threadIdx.x; c < CIN; c += 256) s += qb[(size_t)q * CIN + c];
  red[threadIdx.x] = s;
  __syncthreads();
  for (int st = 128; st > 0; st >>= 1) {
    if (threadIdx.x < st) red[threadIdx.x] += red[threadIdx.x + st];
    __syncthreads();
  }
  if (threadIdx.x == 0) qbits[q] = red[0];
}

// ---------------- decode stage 1: h2 = relu(relu(relu([qe,pe])@W1^T+b1)@W2^T+b2) ----------------
__global__ __launch_bounds__(32) void k_decode1(const bf16* __restrict__ embq,
                                                const bf16* __restrict__ embp,
                                                const bf16* __restrict__ W1b,
                                                const float* __restrict__ b1,
                                                const bf16* __restrict__ W2b,
                                                const float* __restrict__ b2,
                                                bf16* __restrict__ H2out) {
  __shared__ __align__(16) bf16 sh1[16 * 32];
  int t = blockIdx.x;                  // pair tile: pairs [t*16, t*16+16)
  int q = t >> 4, pbase = (t & 15) << 4;
  int lane = threadIdx.x, hk = lane >> 4, ml = lane & 15;
  v8f acc0, acc1;
  float bb0 = b1[ml], bb1 = b1[16 + ml];
#pragma unroll
  for (int r = 0; r < 8; ++r) { acc0[r] = bb0; acc1[r] = bb1; }
  for (int kc = 0; kc < 2 * H1; kc += 32) {
    v16bf a;
    if (kc < H1) a = a_frag(embq + q * H1 + kc, hk);            // same row for all m
    else         a = a_frag(embp + (pbase + ml) * H1 + (kc - H1), hk);
    v16bf w0 = b_frag(W1b + (size_t)ml * (2 * H1) + kc, hk);
    v16bf w1 = b_frag(W1b + (size_t)(16 + ml) * (2 * H1) + kc, hk);
    acc0 = wmma_bf16f32(a, w0, acc0);
    acc1 = wmma_bf16f32(a, w1, acc1);
  }
#pragma unroll
  for (int r = 0; r < 8; ++r) {
    int m = r + hk * 8;
    sh1[m * 32 + ml]      = (bf16)fmaxf(acc0[r], 0.f);
    sh1[m * 32 + 16 + ml] = (bf16)fmaxf(acc1[r], 0.f);
  }
  __syncthreads();
  v16bf a2 = a_frag(sh1 + ml * 32, hk);
  v8f o0, o1;
  float c0 = b2[ml], c1 = b2[16 + ml];
#pragma unroll
  for (int r = 0; r < 8; ++r) { o0[r] = c0; o1[r] = c1; }
  o0 = wmma_bf16f32(a2, b_frag(W2b + (size_t)ml * H2, hk), o0);
  o1 = wmma_bf16f32(a2, b_frag(W2b + (size_t)(16 + ml) * H2, hk), o1);
#pragma unroll
  for (int r = 0; r < 8; ++r) {
    int m = r + hk * 8;
    H2out[(size_t)(t * 16 + m) * H2 + ml]      = (bf16)fmaxf(o0[r], 0.f);
    H2out[(size_t)(t * 16 + m) * H2 + 16 + ml] = (bf16)fmaxf(o1[r], 0.f);
  }
}

// ---------------- decode stage 2: fused w-GEMM + masked bloom reduction ----------------
__global__ __launch_bounds__(512) void k_decode2(const bf16* __restrict__ H2m,
                                                 const bf16* __restrict__ W3b,
                                                 const float* __restrict__ qbloom,
                                                 const float* __restrict__ pbloom,
                                                 const float* __restrict__ qbits,
                                                 float* __restrict__ desc) {
  __shared__ float sacc[16];
  int t = blockIdx.x;
  int q = t >> 4, pbase = (t & 15) << 4;
  int tid = threadIdx.x;
  int lane = tid & 31, wave = tid >> 5;
  int hk = lane >> 4, ml = lane & 15;
  if (tid < 16) sacc[tid] = 0.f;
  __syncthreads();
  v16bf a = a_frag(H2m + (size_t)(t * 16 + ml) * H2, hk);
  float dacc[8];
#pragma unroll
  for (int r = 0; r < 8; ++r) dacc[r] = 0.f;
  const float* qrow = qbloom + (size_t)q * COUT;
  const float* prow = pbloom + (size_t)pbase * COUT;
  for (int nt = wave; nt < COUT / 16; nt += 16) {
    int n0 = nt << 4;
    v16bf w = b_frag(W3b + (size_t)(n0 + ml) * H2, hk);
    v8f z = {};
    v8f d = wmma_bf16f32(a, w, z);
    int c = n0 + ml;
    float qv = qrow[c];
#pragma unroll
    for (int r = 0; r < 8; ++r) {
      int m = r + hk * 8;
      float pv = prow[(size_t)m * COUT + c];
      float x = d[r];
      float lw = (x >= 0.f ? x : 0.125f * x) + 1.f;
      dacc[r] += qv * pv * lw;
    }
  }
#pragma unroll
  for (int r = 0; r < 8; ++r) {
    float s = dacc[r];
    s += __shfl_xor(s, 1);
    s += __shfl_xor(s, 2);
    s += __shfl_xor(s, 4);
    s += __shfl_xor(s, 8);
    if (ml == 0) atomicAdd(&sacc[r + hk * 8], s);
  }
  __syncthreads();
  if (tid < 16) desc[t * 16 + tid] = sacc[tid] / qbits[q];
}

// ---------------- finalize: normalize + geo fusion ----------------
__global__ __launch_bounds__(256) void k_final(const float* __restrict__ desc,
                                               const float* __restrict__ qloc,
                                               const float* __restrict__ ploc,
                                               const float* __restrict__ sa,
                                               const float* __restrict__ sb,
                                               const float* __restrict__ sc,
                                               const float* __restrict__ sd,
                                               float* __restrict__ out) {
  __shared__ float red[256];
  int q = blockIdx.x, p = threadIdx.x;
  float v = desc[q * BP + p];
  red[p] = v;
  __syncthreads();
  for (int s = 128; s > 0; s >>= 1) { if (p < s) red[p] += red[p + s]; __syncthreads(); }
  float mean = red[0] * (1.f / BP);
  __syncthreads();
  float dv = v - mean;
  red[p] = dv * dv;
  __syncthreads();
  for (int s = 128; s > 0; s >>= 1) { if (p < s) red[p] += red[p + s]; __syncthreads(); }
  float stdv = sqrtf(red[0] / (float)(BP - 1));
  float z = dv / (stdv + 1e-6f);
  float dx = qloc[q * 2] - ploc[p * 2];
  float dy = qloc[q * 2 + 1] - ploc[p * 2 + 1];
  float dist = sqrtf(dx * dx + dy * dy);
  float dist_sim = -logf(dist + 1.f);
  float A = sa[0], B = sb[0], C = sc[0], D = sd[0];
  float sg = 1.f / (1.f + expf(-(A * z + B)));
  out[q * BP + p] = (C - sg) * (dist_sim - D);
}

extern "C" void kernel_launch(void* const* d_in, const int* in_sizes, int n_in,
                              void* d_out, int out_size, void* d_ws, size_t ws_size,
                              hipStream_t stream) {
  (void)in_sizes; (void)n_in; (void)out_size; (void)ws_size;
  const float* query_x  = (const float*)d_in[0];
  const float* query_loc= (const float*)d_in[1];
  const float* poi_x    = (const float*)d_in[2];
  const float* poi_loc  = (const float*)d_in[3];
  const float* enc_w    = (const float*)d_in[4];
  const float* conv_w1  = (const float*)d_in[5];
  const float* conv_b1  = (const float*)d_in[6];
  const float* conv_w2  = (const float*)d_in[7];
  const float* conv_b2  = (const float*)d_in[8];
  const float* conv_w3  = (const float*)d_in[9];
  const float* conv_b3  = (const float*)d_in[10];
  const float* dec_w1   = (const float*)d_in[11];
  const float* dec_b1   = (const float*)d_in[12];
  const float* dec_w2   = (const float*)d_in[13];
  const float* dec_b2   = (const float*)d_in[14];
  const float* dec_w3   = (const float*)d_in[15];
  const float* s_a = (const float*)d_in[16];
  const float* s_b = (const float*)d_in[17];
  const float* s_c = (const float*)d_in[18];
  const float* s_d = (const float*)d_in[19];
  float* out = (float*)d_out;

  char* ws = (char*)d_ws;
  size_t off = 0;
  auto alloc = [&](size_t bytes) -> void* {
    void* p = ws + off;
    off = (off + bytes + 255) & ~(size_t)255;
    return p;
  };
  bf16*  enc_wb = (bf16*)alloc((size_t)H1 * CIN * 2);
  bf16*  convwb = (bf16*)alloc((size_t)3 * 5 * H1 * H1 * 2);
  bf16*  w1b    = (bf16*)alloc((size_t)H2 * 2 * H1 * 2);
  bf16*  w2b    = (bf16*)alloc((size_t)H2 * H2 * 2);
  bf16*  w3b    = (bf16*)alloc((size_t)COUT * H2 * 2);
  float* Y      = (float*)alloc((size_t)(BQ + BP) * LTOK * H1 * 4);
  bf16*  embq   = (bf16*)alloc((size_t)BQ * H1 * 2);
  bf16*  embp   = (bf16*)alloc((size_t)BP * H1 * 2);
  float* qbloom = (float*)alloc((size_t)BQ * CIN * 4);
  float* pbloom = (float*)alloc((size_t)BP * CIN * 4);
  float* qbits  = (float*)alloc((size_t)BQ * 4);
  bf16*  h2     = (bf16*)alloc((size_t)BQ * BP * H2 * 2);
  float* desc   = (float*)alloc((size_t)BQ * BP * 4);

  // weight preconversion to bf16
  k_f2bf <<<(H1 * CIN + 255) / 256, 256, 0, stream>>>(enc_w, enc_wb, H1 * CIN);
  k_convw<<<(5 * H1 * H1 + 255) / 256, 256, 0, stream>>>(conv_w1, convwb);
  k_convw<<<(5 * H1 * H1 + 255) / 256, 256, 0, stream>>>(conv_w2, convwb + 5 * H1 * H1);
  k_convw<<<(5 * H1 * H1 + 255) / 256, 256, 0, stream>>>(conv_w3, convwb + 2 * 5 * H1 * H1);
  k_f2bf <<<(H2 * 2 * H1 + 255) / 256, 256, 0, stream>>>(dec_w1, w1b, H2 * 2 * H1);
  k_f2bf <<<(H2 * H2 + 255) / 256, 256, 0, stream>>>(dec_w2, w2b, H2 * H2);
  k_f2bf <<<(COUT * H2 + 255) / 256, 256, 0, stream>>>(dec_w3, w3b, COUT * H2);

  // encoder GEMM (query rows then poi rows)
  k_encode<<<BQ * LTOK / 16, 512, 0, stream>>>(query_x, enc_wb, Y);
  k_encode<<<BP * LTOK / 16, 512, 0, stream>>>(poi_x, enc_wb, Y + (size_t)BQ * LTOK * H1);

  // conv block + residual + mean -> relu'd embeddings (bf16)
  k_conv<<<BQ + BP, 512, 0, stream>>>(Y, convwb, conv_b1, conv_b2, conv_b3, embq, embp);

  // blooms + q_bits
  k_bloom<<<BQ * CIN / 256, 256, 0, stream>>>(query_x, qbloom);
  k_bloom<<<BP * CIN / 256, 256, 0, stream>>>(poi_x, pbloom);
  k_qbits<<<BQ, 256, 0, stream>>>(qbloom, qbits);

  // pairwise decode
  k_decode1<<<(BQ * BP) / 16, 32, 0, stream>>>(embq, embp, w1b, dec_b1, w2b, dec_b2, h2);
  k_decode2<<<(BQ * BP) / 16, 512, 0, stream>>>(h2, w3b, qbloom, pbloom, qbits, desc);

  // finalize
  k_final<<<BQ, BP, 0, stream>>>(desc, query_loc, poi_loc, s_a, s_b, s_c, s_d, out);
}
